// ImageTransformerDenoiserModelV2_35777077575815
// MI455X (gfx1250) — compile-verified, matrix operand order checked by reading, not run on hardware
//
#include <hip/hip_runtime.h>
#include <math.h>

// ---------------------------------------------------------------------------
// Types for CDNA5 WMMA (wave32, v_wmma_f32_16x16x32_bf16)
// ---------------------------------------------------------------------------
typedef __attribute__((ext_vector_type(16))) __bf16 v16bf;
typedef __attribute__((ext_vector_type(8)))  float  v8f;
typedef __attribute__((ext_vector_type(4)))  int    v4i;

typedef __attribute__((address_space(1))) v4i gv4i;   // global int4
typedef __attribute__((address_space(3))) v4i lv4i;   // LDS int4

#if __has_builtin(__builtin_amdgcn_global_load_async_to_lds_b128)
#define HAVE_ASYNC_LDS 1
#endif

#if __has_builtin(__builtin_amdgcn_s_wait_asynccnt)
#define WAIT_ASYNC(n) __builtin_amdgcn_s_wait_asynccnt(n)
#elif defined(HAVE_ASYNC_LDS)
#define WAIT_ASYNC(n) asm volatile("s_wait_asynccnt " #n ::: "memory")
#else
#define WAIT_ASYNC(n)
#endif

__device__ __forceinline__ __bf16 f2bf(float f) {
    unsigned u = __builtin_bit_cast(unsigned, f);
    unsigned r = (u + 0x7FFFu + ((u >> 16) & 1u)) >> 16;
    return __builtin_bit_cast(__bf16, (unsigned short)r);
}
__device__ __forceinline__ float bf2f(__bf16 b) {
    unsigned u = ((unsigned)__builtin_bit_cast(unsigned short, b)) << 16;
    return __builtin_bit_cast(float, u);
}

union Frag { v16bf v; v4i i[2]; };

// Load one 16x32 bf16 fragment (A-layout; identical pattern serves B when the
// B operand is stored [N][K] row-major).  ISA 7.12.2: lanes 0-15 hold row
// M=lane, K = {0..7, 16..23}; lanes 16-31 hold row M=lane-16, K = {8..15, 24..31}.
__device__ __forceinline__ v16bf load_frag(const __bf16* base, int ld, int lane) {
    const __bf16* p = base + (lane & 15) * ld + ((lane >> 4) << 3);
    Frag f;
    f.i[0] = *(const v4i*)(p);
    f.i[1] = *(const v4i*)(p + 16);
    return f.v;
}

__device__ __forceinline__ v8f wmma_bf16(v16bf a, v16bf b, v8f c) {
    return __builtin_amdgcn_wmma_f32_16x16x32_bf16(false, a, false, b, (short)0, c, false, false);
}

// Async 16-byte global -> LDS copy (per-lane addresses), ASYNCcnt-tracked.
__device__ __forceinline__ void cp16_async(const __bf16* g, __bf16* l) {
#ifdef HAVE_ASYNC_LDS
    __builtin_amdgcn_global_load_async_to_lds_b128(
        (gv4i*)(uintptr_t)(const void*)g,
        (lv4i*)(unsigned)(uintptr_t)(void*)l,
        0, 0);
#else
    *(v4i*)l = *(const v4i*)g;
#endif
}

// ---------------------------------------------------------------------------
// Problem constants
// ---------------------------------------------------------------------------
#define NB   4
#define HH   128
#define WW_  128
#define DM   768
#define NH   12
#define DH   64
#define DF   2048
#define WS   8
#define SHIFT 4
#define MROWS (NB * HH * WW_)      // 65536
#define EPSF 1e-6f

// ---------------------------------------------------------------------------
// fp32 -> bf16 cast
// ---------------------------------------------------------------------------
__global__ void cvt_bf16_kernel(const float* __restrict__ src, __bf16* __restrict__ dst, int n) {
    int i = blockIdx.x * 256 + threadIdx.x;
    if (i < n) dst[i] = f2bf(src[i]);
}

// ---------------------------------------------------------------------------
// ada scales: s[n][d] = 1 + cond[n,:] . w[d,:]   (two weight matrices)
// ---------------------------------------------------------------------------
__global__ void ada_scale_kernel(const float* __restrict__ cond,
                                 const float* __restrict__ w1, const float* __restrict__ w2,
                                 float* __restrict__ s1, float* __restrict__ s2) {
    int t = blockIdx.x * 256 + threadIdx.x;
    if (t >= 2 * NB * DM) return;
    int d = t % DM;
    int n = (t / DM) & (NB - 1);
    int which = t / (NB * DM);
    const float* w = which ? w2 : w1;
    float acc = 0.f;
    for (int k = 0; k < DM; ++k) acc += cond[n * DM + k] * w[d * DM + k];
    (which ? s2 : s1)[n * DM + d] = acc + 1.0f;
}

// ---------------------------------------------------------------------------
// ada-RMS norm + cast to bf16.  One 256-thread block per 768-wide row.
// ---------------------------------------------------------------------------
__global__ __launch_bounds__(256) void rms_cast_kernel(const float* __restrict__ x,
                                                       const float* __restrict__ s,
                                                       __bf16* __restrict__ out) {
    int m = blockIdx.x, tid = threadIdx.x;
    int n = m >> 14;                      // 16384 rows per batch element
    __shared__ float red[256];
    size_t base = (size_t)m * DM;
    float xv[3]; float sq = 0.f;
    #pragma unroll
    for (int j = 0; j < 3; ++j) { float v = x[base + j * 256 + tid]; xv[j] = v; sq += v * v; }
    red[tid] = sq; __syncthreads();
    for (int off = 128; off; off >>= 1) { if (tid < off) red[tid] += red[tid + off]; __syncthreads(); }
    float inv = rsqrtf(red[0] * (1.0f / DM) + EPSF);
    #pragma unroll
    for (int j = 0; j < 3; ++j) {
        int d = j * 256 + tid;
        out[base + d] = f2bf(xv[j] * s[n * DM + d] * inv);
    }
}

// ---------------------------------------------------------------------------
// Generic bf16 WMMA GEMM:  C[M,N] = A[M,K] (row-major bf16) * B[N,K]^T
// block tile 128x128, 8 waves of 32x64.  32-wide K tiles double-buffered in
// LDS via GLOBAL_LOAD_ASYNC_TO_LDS_B128 (ASYNCcnt) so the next tile streams
// while WMMAs consume the current one.
// RES=false: write bf16.  RES=true: write fp32 = residual + acc.
// ---------------------------------------------------------------------------
template<bool RES>
__global__ __launch_bounds__(256) void gemm_kernel(const __bf16* __restrict__ A,
                                                   const __bf16* __restrict__ B,
                                                   __bf16* __restrict__ outb,
                                                   float* __restrict__ outf,
                                                   const float* __restrict__ res,
                                                   int N, int K) {
    __shared__ __bf16 As[2][128 * 40];
    __shared__ __bf16 Bs[2][128 * 40];
    const int tid = threadIdx.x, lane = tid & 31, wid = tid >> 5;
    const int wm = (wid >> 1) * 32;       // wave row offset within block tile
    const int wn = (wid & 1) * 64;        // wave col offset within block tile
    const long rowA0 = (long)blockIdx.y * 128;
    const long rowB0 = (long)blockIdx.x * 128;

    // per-thread staging slots: two 16B chunks per operand per k-tile
    const int i0 = tid * 8;               // element 0..2047
    const int r0 = i0 >> 5, c0 = i0 & 31;
    const int r1 = (i0 + 2048) >> 5;      // second chunk, same column

    v8f acc[2][4];
    v8f z = {0.f, 0.f, 0.f, 0.f, 0.f, 0.f, 0.f, 0.f};
    #pragma unroll
    for (int i = 0; i < 2; ++i)
        #pragma unroll
        for (int j = 0; j < 4; ++j) acc[i][j] = z;

    auto issue = [&](int kt, int buf) {
        cp16_async(A + (rowA0 + r0) * K + kt + c0, As[buf] + r0 * 40 + c0);
        cp16_async(A + (rowA0 + r1) * K + kt + c0, As[buf] + r1 * 40 + c0);
        cp16_async(B + (rowB0 + r0) * K + kt + c0, Bs[buf] + r0 * 40 + c0);
        cp16_async(B + (rowB0 + r1) * K + kt + c0, Bs[buf] + r1 * 40 + c0);
    };

    const int nk = K >> 5;
    issue(0, 0);
    for (int i = 0; i < nk; ++i) {
        const int cur = i & 1;
        if (i + 1 < nk) {
            issue((i + 1) << 5, cur ^ 1);   // prefetch next tile into other buffer
            WAIT_ASYNC(4);                  // current tile done; prefetch in flight
        } else {
            WAIT_ASYNC(0);
        }
        __syncthreads();
        v16bf af[2], bfr[4];
        af[0] = load_frag(As[cur] + (wm +  0) * 40, 40, lane);
        af[1] = load_frag(As[cur] + (wm + 16) * 40, 40, lane);
        #pragma unroll
        for (int tn = 0; tn < 4; ++tn) bfr[tn] = load_frag(Bs[cur] + (wn + tn * 16) * 40, 40, lane);
        #pragma unroll
        for (int tm = 0; tm < 2; ++tm)
            #pragma unroll
            for (int tn = 0; tn < 4; ++tn)
                acc[tm][tn] = wmma_bf16(af[tm], bfr[tn], acc[tm][tn]);
        __syncthreads();                    // protect WAR before next prefetch lands
    }

    const int half = lane >> 4, col0 = lane & 15;
    #pragma unroll
    for (int tm = 0; tm < 2; ++tm)
        #pragma unroll
        for (int tn = 0; tn < 4; ++tn)
            #pragma unroll
            for (int v = 0; v < 8; ++v) {
                long row = rowA0 + wm + tm * 16 + v + 8 * half;
                long col = rowB0 + wn + tn * 16 + col0;
                long o = row * (long)N + col;
                if (RES) outf[o] = res[o] + acc[tm][tn][v];
                else     outb[o] = f2bf(acc[tm][tn][v]);
            }
}

// ---------------------------------------------------------------------------
// QK L2-norm (sum of squares, * sqrt(attn_scale)) + RoPE, in place (bf16 qkv)
// One 64-thread block per (token row m, head).
// ---------------------------------------------------------------------------
__global__ __launch_bounds__(64) void qknorm_rope_kernel(__bf16* __restrict__ qkv,
                                                         const float* __restrict__ pos,
                                                         const float* __restrict__ attn_scale,
                                                         const float* __restrict__ rf /* 12x8 */) {
    int m = blockIdx.x, head = blockIdx.y, e = threadIdx.x;
    __shared__ float qs_[64], ks_[64], red[4];
    size_t base = (size_t)m * (3 * DM) + head * DH;
    float qv = bf2f(qkv[base + e]);
    float kv = bf2f(qkv[base + DM + e]);
    float sq = qv * qv, sk = kv * kv;
    #pragma unroll
    for (int off = 16; off; off >>= 1) {
        sq += __shfl_xor(sq, off, 32);
        sk += __shfl_xor(sk, off, 32);
    }
    if ((e & 31) == 0) { red[(e >> 5) * 2] = sq; red[(e >> 5) * 2 + 1] = sk; }
    __syncthreads();
    float sumq = red[0] + red[2], sumk = red[1] + red[3];
    float ss = sqrtf(attn_scale[head]);
    float qn = ss * rsqrtf(sumq + EPSF);
    float kn = ss * rsqrtf(sumk + EPSF);
    qs_[e] = qv * qn; ks_[e] = kv * kn;
    __syncthreads();
    float oq = qs_[e], ok = ks_[e];
    if (e < 32) {
        int j = e & 15;
        float ph = pos[(size_t)m * 2 + 0], pw = pos[(size_t)m * 2 + 1];
        float th = (j < 8) ? ph * rf[head * 8 + j] : pw * rf[head * 8 + (j - 8)];
        float c = __cosf(th), s = __sinf(th);
        if (e < 16) { oq = qs_[e] * c - qs_[e + 16] * s; ok = ks_[e] * c - ks_[e + 16] * s; }
        else        { oq = qs_[e] * c + qs_[e - 16] * s; ok = ks_[e] * c + ks_[e - 16] * s; }
    }
    qkv[base + e]      = f2bf(oq);
    qkv[base + DM + e] = f2bf(ok);
}

// ---------------------------------------------------------------------------
// Shifted-window attention: one 4-wave block per (window, head).
// 64x64 scores via WMMA, analytic shift mask, fp32 softmax via shfl over the
// 16-lane column groups, P staged bf16 in LDS, P*V via WMMA, scatter with
// inverse roll.
// ---------------------------------------------------------------------------
__global__ __launch_bounds__(128) void attn_kernel(const __bf16* __restrict__ qkv,
                                                   __bf16* __restrict__ o) {
    const int widx = blockIdx.x;            // 0..255 = wh*16+ww
    const int head = blockIdx.y, b = blockIdx.z;
    const int wh = widx >> 4, ww = widx & 15;
    __shared__ __bf16 qs[64 * 72];
    __shared__ __bf16 ks[64 * 72];
    __shared__ __bf16 vT[64 * 72];          // [dim][token]
    __shared__ __bf16 ps[64 * 72];          // probabilities
    const int tid = threadIdx.x, lane = tid & 31, wv = tid >> 5;

    // ---- stage q, k (row-major) and v (transposed) for this window/head ----
    {
        int t = tid >> 1, part = tid & 1;
        int qh = t >> 3, qw = t & 7;
        int hh = (wh * WS + qh - SHIFT) & (HH - 1);
        int wp = (ww * WS + qw - SHIFT) & (WW_ - 1);
        size_t rowbase = ((size_t)(b * HH + hh) * WW_ + wp) * (3 * DM) + head * DH;
        #pragma unroll
        for (int c = 0; c < 4; ++c) {
            int e0 = part * 32 + c * 8;
            *(v4i*)(qs + t * 72 + e0) = *(const v4i*)(qkv + rowbase + e0);
            *(v4i*)(ks + t * 72 + e0) = *(const v4i*)(qkv + rowbase + DM + e0);
            const __bf16* vp = qkv + rowbase + 2 * DM + e0;
            #pragma unroll
            for (int u = 0; u < 8; ++u) vT[(e0 + u) * 72 + t] = vp[u];
        }
    }
    __syncthreads();

    // ---- scores: strip rows [wv*16, wv*16+16) x all 64 cols -----------------
    v8f z = {0.f, 0.f, 0.f, 0.f, 0.f, 0.f, 0.f, 0.f};
    v8f st[4] = {z, z, z, z};
    #pragma unroll
    for (int kc = 0; kc < 2; ++kc) {
        v16bf a = load_frag(qs + (wv * 16) * 72 + kc * 32, 72, lane);
        #pragma unroll
        for (int tn = 0; tn < 4; ++tn) {
            v16bf bb = load_frag(ks + (tn * 16) * 72 + kc * 32, 72, lane);
            st[tn] = wmma_bf16(a, bb, st[tn]);
        }
    }

    // ---- mask + softmax -----------------------------------------------------
    const bool top = (wh == 0), left = (ww == 0);
    const int half = lane >> 4, col0 = lane & 15;
    bool kaA[4], klA[4];
    #pragma unroll
    for (int tn = 0; tn < 4; ++tn) {
        int tk = tn * 16 + col0;
        kaA[tn] = (tk >> 3) < SHIFT;
        klA[tn] = (tk & 7)  < SHIFT;
    }
    float p_[4][8];
    #pragma unroll
    for (int v = 0; v < 8; ++v) {
        int tq = wv * 16 + v + 8 * half;
        bool qa = (tq >> 3) < SHIFT, ql = (tq & 7) < SHIFT;
        float sv[4], mx = -3.0e38f;
        #pragma unroll
        for (int tn = 0; tn < 4; ++tn) {
            bool mok = (!left && !top) ||
                       ( left &&  top && (ql == klA[tn]) && (qa == kaA[tn])) ||
                       ( left && !top && (ql == klA[tn])) ||
                       (!left &&  top && (qa == kaA[tn]));
            sv[tn] = mok ? st[tn][v] : -1.0e30f;
            mx = fmaxf(mx, sv[tn]);
        }
        #pragma unroll
        for (int off = 1; off < 16; off <<= 1) mx = fmaxf(mx, __shfl_xor(mx, off, 32));
        float sum = 0.f;
        #pragma unroll
        for (int tn = 0; tn < 4; ++tn) { sv[tn] = __expf(sv[tn] - mx); sum += sv[tn]; }
        #pragma unroll
        for (int off = 1; off < 16; off <<= 1) sum += __shfl_xor(sum, off, 32);
        float inv = 1.0f / sum;
        #pragma unroll
        for (int tn = 0; tn < 4; ++tn) p_[tn][v] = sv[tn] * inv;
    }
    #pragma unroll
    for (int v = 0; v < 8; ++v) {
        int row = wv * 16 + v + 8 * half;
        #pragma unroll
        for (int tn = 0; tn < 4; ++tn)
            ps[row * 72 + tn * 16 + col0] = f2bf(p_[tn][v]);
    }
    __syncthreads();

    // ---- O strip = P(16x64) @ V(64x64) --------------------------------------
    v8f ot[4] = {z, z, z, z};
    #pragma unroll
    for (int kc = 0; kc < 2; ++kc) {
        v16bf a = load_frag(ps + (wv * 16) * 72 + kc * 32, 72, lane);
        #pragma unroll
        for (int tn = 0; tn < 4; ++tn) {
            v16bf bb = load_frag(vT + (tn * 16) * 72 + kc * 32, 72, lane);
            ot[tn] = wmma_bf16(a, bb, ot[tn]);
        }
    }

    // ---- scatter with inverse roll -----------------------------------------
    #pragma unroll
    for (int v = 0; v < 8; ++v) {
        int tq = wv * 16 + v + 8 * half;
        int hh = (wh * WS + (tq >> 3) - SHIFT) & (HH - 1);
        int wp = (ww * WS + (tq & 7)  - SHIFT) & (WW_ - 1);
        size_t obase = ((size_t)(b * HH + hh) * WW_ + wp) * DM + head * DH;
        #pragma unroll
        for (int tn = 0; tn < 4; ++tn)
            o[obase + tn * 16 + col0] = f2bf(ot[tn][v]);
    }
}

// ---------------------------------------------------------------------------
// GEGLU gate: gated[m,j] = u[m,j] * gelu(u[m,2048+j])   (exact erf gelu)
// ---------------------------------------------------------------------------
__global__ __launch_bounds__(256) void gate_kernel(const __bf16* __restrict__ u,
                                                   __bf16* __restrict__ g) {
    size_t i = (size_t)blockIdx.x * 256 + threadIdx.x;   // 65536*2048 total
    size_t mrow = i >> 11;
    int j = (int)(i & 2047);
    float a  = bf2f(u[mrow * (2 * DF) + j]);
    float gg = bf2f(u[mrow * (2 * DF) + DF + j]);
    float ge = 0.5f * gg * (1.0f + erff(gg * 0.70710678118654752f));
    g[i] = f2bf(a * ge);
}

// ---------------------------------------------------------------------------
// Host-side launch
// ---------------------------------------------------------------------------
extern "C" void kernel_launch(void* const* d_in, const int* in_sizes, int n_in,
                              void* d_out, int out_size, void* d_ws, size_t ws_size,
                              hipStream_t stream) {
    const float* x          = (const float*)d_in[0];
    const float* pos        = (const float*)d_in[1];
    const float* cond       = (const float*)d_in[2];
    const float* ada1_w     = (const float*)d_in[3];
    const float* qkv_w      = (const float*)d_in[4];
    const float* attn_scale = (const float*)d_in[5];
    const float* rope_freqs = (const float*)d_in[6];
    const float* out_w      = (const float*)d_in[7];
    const float* ada2_w     = (const float*)d_in[8];
    const float* ff_up_w    = (const float*)d_in[9];
    const float* ff_down_w  = (const float*)d_in[10];
    float* out = (float*)d_out;

    char* wsp = (char*)d_ws;
    auto alloc = [&](size_t bytes) {
        char* p = wsp;
        wsp += (bytes + 255) & ~(size_t)255;
        return p;
    };
    float*  s1    = (float*)alloc((size_t)NB * DM * 4);
    float*  s2    = (float*)alloc((size_t)NB * DM * 4);
    __bf16* wqkv  = (__bf16*)alloc((size_t)3 * DM * DM * 2);
    __bf16* wout  = (__bf16*)alloc((size_t)DM * DM * 2);
    __bf16* wup   = (__bf16*)alloc((size_t)2 * DF * DM * 2);
    __bf16* wdown = (__bf16*)alloc((size_t)DM * DF * 2);
    __bf16* hb    = (__bf16*)alloc((size_t)MROWS * DM * 2);
    // phase-overlapped arena: {qkv, o} then {u, gated}
    char* arena   = alloc((size_t)MROWS * (2 * DF + DF) * 2);   // 805 MB
    __bf16* qkvb  = (__bf16*)arena;
    __bf16* ob    = (__bf16*)(arena + (size_t)MROWS * 3 * DM * 2);
    __bf16* ub    = (__bf16*)arena;
    __bf16* gated = (__bf16*)(arena + (size_t)MROWS * 2 * DF * 2);

    // 1) weights -> bf16
    {
        int n;
        n = 3 * DM * DM; cvt_bf16_kernel<<<(n + 255) / 256, 256, 0, stream>>>(qkv_w, wqkv, n);
        n = DM * DM;     cvt_bf16_kernel<<<(n + 255) / 256, 256, 0, stream>>>(out_w, wout, n);
        n = 2 * DF * DM; cvt_bf16_kernel<<<(n + 255) / 256, 256, 0, stream>>>(ff_up_w, wup, n);
        n = DM * DF;     cvt_bf16_kernel<<<(n + 255) / 256, 256, 0, stream>>>(ff_down_w, wdown, n);
    }
    // 2) ada scales
    ada_scale_kernel<<<(2 * NB * DM + 255) / 256, 256, 0, stream>>>(cond, ada1_w, ada2_w, s1, s2);
    // 3) h = ada_rms(x, s1) -> bf16
    rms_cast_kernel<<<MROWS, 256, 0, stream>>>(x, s1, hb);
    // 4) qkv = h @ qkv_w.T -> bf16
    gemm_kernel<false><<<dim3((3 * DM) / 128, MROWS / 128), 256, 0, stream>>>(
        hb, wqkv, qkvb, nullptr, nullptr, 3 * DM, DM);
    // 5) qk norm + rope (in place)
    qknorm_rope_kernel<<<dim3(MROWS, NH), 64, 0, stream>>>(qkvb, pos, attn_scale, rope_freqs);
    // 6) windowed attention
    attn_kernel<<<dim3((HH / WS) * (WW_ / WS), NH, NB), 128, 0, stream>>>(qkvb, ob);
    // 7) x1 = x + o @ out_w.T  (fp32 into d_out)
    gemm_kernel<true><<<dim3(DM / 128, MROWS / 128), 256, 0, stream>>>(
        ob, wout, nullptr, out, x, DM, DM);
    // 8) h = ada_rms(x1, s2) -> bf16
    rms_cast_kernel<<<MROWS, 256, 0, stream>>>(out, s2, hb);
    // 9) u = h @ ff_up_w.T -> bf16
    gemm_kernel<false><<<dim3((2 * DF) / 128, MROWS / 128), 256, 0, stream>>>(
        hb, wup, ub, nullptr, nullptr, 2 * DF, DM);
    // 10) gated = a * gelu(g)
    gate_kernel<<<(int)(((size_t)MROWS * DF) / 256), 256, 0, stream>>>(ub, gated);
    // 11) out = x1 + gated @ ff_down_w.T
    gemm_kernel<true><<<dim3(DM / 128, MROWS / 128), 256, 0, stream>>>(
        gated, wdown, nullptr, out, out, DM, DF);
}
